// Word2Vec_730144441175
// MI455X (gfx1250) — matrix operand order repaired and checked by reading, not compiled, not run on hardware
//
#include <hip/hip_runtime.h>

// ---------------------------------------------------------------------------
// CBOW word2vec loss, fused:
//   K1: ctx_vec = mean(emb[context_ids]) (f32 + bf16 copy)
//   K2: weight -> bf16 (padded to multiple of 16 rows, pad rows zeroed)
//   K3: WMMA bf16 GEMM fused with online logsumexp. B tiles (16x128 bf16,
//       4KB) staged into double-buffered LDS by the Tensor Data Mover
//       (tensor_load_to_lds, TENSORcnt) when available, else by manual
//       global->VGPR->ds_store staging. 256 rows/WG -> 64 WMMAs per tile.
//   K4: merge partials, add center-token dot (f32), loss = logZ - logit_c
// ---------------------------------------------------------------------------

#define VOCAB   50257
#define DMODEL  128
#define BATCH   4096
#define CTXW    10
#define VPAD    50272           // 3142 * 16
#define NTILES  3142            // VPAD / 16
#define NSPLIT  32
#define TPS     99              // ceil(NTILES / NSPLIT)

#define ROWS_PER_WG 256         // 8 waves * 32 rows
#define LDS_ROW 136             // halves per LDS row (272B: 16B aligned, 4-bank rotate)
#define LDS_BUF (16 * LDS_ROW)  // halves per buffer (4352 B)
#define PFDIST  8               // prefetch distance in tiles

typedef __attribute__((ext_vector_type(16))) __bf16 v16bf;
typedef __attribute__((ext_vector_type(8)))  __bf16 v8bf;
typedef __attribute__((ext_vector_type(8)))  float  v8f;

#if defined(__has_builtin)
#if __has_builtin(__builtin_amdgcn_tensor_load_to_lds) && \
    __has_builtin(__builtin_amdgcn_s_wait_tensorcnt)
#define USE_TDM 1
#endif
#endif
#ifndef USE_TDM
#define USE_TDM 0
#endif

#if USE_TDM
typedef unsigned int u32x4 __attribute__((ext_vector_type(4)));
typedef int          i32x4 __attribute__((ext_vector_type(4)));
typedef int          i32x8 __attribute__((ext_vector_type(8)));

// TDM: DMA one 16x128-bf16 weight tile global->LDS, writing 16B of LDS pad
// after every 256B row (reproduces the 272B conflict-free row stride).
__device__ __forceinline__ void tdm_load_tile(const unsigned short* gsrc,
                                              unsigned lds_off) {
    unsigned long long ga = (unsigned long long)(size_t)gsrc;
    u32x4 g0;
    g0.x = 1u;                                        // count=1, user D#
    g0.y = lds_off;                                   // lds_addr (bytes)
    g0.z = (unsigned)ga;                              // global_addr[31:0]
    g0.w = (unsigned)((ga >> 32) & 0x01FFFFFFu)       // global_addr[56:32]
         | (2u << 30);                                // type = 2 ("image")
    i32x8 g1;
    g1[0] = (int)((1u << 16)                          // data_size = 2 bytes
                | (1u << 20)                          // pad_enable
                | (5u << 22)                          // pad_interval: 64 dw = 256B
                | (3u << 25));                        // pad_amount: 4 dw = 16B
    g1[1] = (int)(128u << 16);                        // tensor_dim0 = 128
    g1[2] = (int)(16u << 16);                         // tensor_dim1 = 16
    g1[3] = (int)(128u << 16);                        // tile_dim0 = 128
    g1[4] = 16;                                       // tile_dim1 = 16
    g1[5] = 128;                                      // tensor_dim0_stride = 128
    g1[6] = 0;
    g1[7] = 0;
    i32x4 z4 = {0, 0, 0, 0};
#if __has_include(<hip/amd_detail/amd_gfx1250_TDM.h>)
    i32x8 z8 = {0, 0, 0, 0, 0, 0, 0, 0};
    __builtin_amdgcn_tensor_load_to_lds(g0, g1, z4, z4, z8, 0);   // clang-23 form
#else
    __builtin_amdgcn_tensor_load_to_lds(g0, g1, z4, z4, 0);       // ROCm 7.2 form
#endif
}
#endif  // USE_TDM

__device__ __forceinline__ unsigned short f32_to_bf16_rne(float f) {
    unsigned int u = __float_as_uint(f);
    unsigned int r = u + 0x7FFFu + ((u >> 16) & 1u);   // round to nearest even
    return (unsigned short)(r >> 16);
}

// ---- K1: gather + mean pool, emit f32 ctx_vec and bf16 copy ---------------
__global__ void __launch_bounds__(256)
k_ctx_mean(const float* __restrict__ emb, const int* __restrict__ ctx_ids,
           float* __restrict__ ctx_f32, unsigned short* __restrict__ ctx_bf16) {
    int idx = blockIdx.x * 256 + threadIdx.x;          // over BATCH*DMODEL
    if (idx >= BATCH * DMODEL) return;
    int b = idx >> 7;
    int d = idx & 127;
    float s = 0.f;
    #pragma unroll
    for (int i = 0; i < CTXW; ++i) {
        int id = ctx_ids[b * CTXW + i];
        s += emb[id * DMODEL + d];
    }
    s *= (1.0f / CTXW);
    ctx_f32[idx]  = s;
    ctx_bf16[idx] = f32_to_bf16_rne(s);
}

// ---- K2: weight f32 -> bf16, zero-pad rows [VOCAB, VPAD) ------------------
__global__ void __launch_bounds__(256)
k_w_bf16(const float* __restrict__ w, unsigned short* __restrict__ wbf) {
    int idx = blockIdx.x * 256 + threadIdx.x;          // over VPAD*DMODEL
    if (idx >= VPAD * DMODEL) return;
    int n = idx >> 7;
    wbf[idx] = (n < VOCAB) ? f32_to_bf16_rne(w[idx]) : (unsigned short)0;
}

// ---- online logsumexp update: rare-rescale form (1 exp on the hot path) ---
__device__ __forceinline__ void lse_update(float x, float& m, float& s) {
    if (x > m) {                       // rare after warm-up (~ln(V) times/row)
        s *= __expf(m - x);
        m = x;
    }
    s += __expf(x - m);
}

// ---- K3: LDS-staged fused WMMA GEMM + online logsumexp --------------------
// grid = (BATCH/256, NSPLIT), block = 256 (8 waves).
__global__ void __launch_bounds__(256)
k_gemm_lse(const unsigned short* __restrict__ ctx_bf16,
           const unsigned short* __restrict__ wbf,
           float2* __restrict__ partials) {
    const int tid    = threadIdx.x;
    const int lane   = tid & 31;
    const int w      = tid >> 5;                       // wave in WG: 0..7
    const int l15    = lane & 15;
    const int hi16   = lane >> 4;                      // 0 | 1
    const int vsplit = blockIdx.y;
    const int rbase  = blockIdx.x * ROWS_PER_WG + w * 32;  // wave's 32 rows

    __shared__ __align__(16) unsigned short smem[2 * LDS_BUF];

    // A fragments for two 16-row tiles (resident for the whole loop).
    const int ka = hi16 << 3;
    v16bf a0[4], a1[4];
    {
        const unsigned short* ab0 = ctx_bf16 + (size_t)(rbase + l15) * DMODEL;
        const unsigned short* ab1 = ab0 + 16 * DMODEL;
        #pragma unroll
        for (int kf = 0; kf < 4; ++kf) {
            const int kb = kf * 32 + ka;
            v8bf lo0 = *(const v8bf*)(ab0 + kb);
            v8bf hi0 = *(const v8bf*)(ab0 + kb + 16);
            a0[kf] = __builtin_shufflevector(lo0, hi0, 0,1,2,3,4,5,6,7,
                                                       8,9,10,11,12,13,14,15);
            v8bf lo1 = *(const v8bf*)(ab1 + kb);
            v8bf hi1 = *(const v8bf*)(ab1 + kb + 16);
            a1[kf] = __builtin_shufflevector(lo1, hi1, 0,1,2,3,4,5,6,7,
                                                       8,9,10,11,12,13,14,15);
        }
    }

    float rm0[8], rs0[8], rm1[8], rs1[8];
    #pragma unroll
    for (int j = 0; j < 8; ++j) {
        rm0[j] = -INFINITY; rs0[j] = 0.f;
        rm1[j] = -INFINITY; rs1[j] = 0.f;
    }

    const int srow   = tid >> 4;                       // 0..15 (weight row)
    const int schunk = tid & 15;                       // 0..15 (8-half chunk)
    const int kbB    = hi16 << 3;                      // B frag: table K order
    int t0 = vsplit * TPS;
    int t1 = t0 + TPS; if (t1 > NTILES) t1 = NTILES;

#if USE_TDM
    const unsigned lds_base = (unsigned)(size_t)&smem[0];
    // Preload first tile via the Tensor Data Mover (wave 0 only; EXEC-free).
    if (w == 0) {
        tdm_load_tile(wbf + (size_t)t0 * 16 * DMODEL, lds_base);
        __builtin_amdgcn_s_wait_tensorcnt(0);
    }
    __syncthreads();
#else
    const int loff = srow * LDS_ROW + schunk * 8;      // LDS halves offset
    *(uint4*)&smem[loff] =
        *(const uint4*)(wbf + (size_t)(t0 * 16 + srow) * DMODEL + schunk * 8);
    __syncthreads();
#endif

    for (int t = t0; t < t1; ++t) {
        const int  cur = (t - t0) & 1;
        const bool hn  = (t + 1 < t1);

#if USE_TDM
        // Kick next tile's DMA now; it runs concurrently with the WMMA chain.
        if (w == 0 && hn)
            tdm_load_tile(wbf + (size_t)(t + 1) * 16 * DMODEL,
                          lds_base + (unsigned)((cur ^ 1) * LDS_BUF * 2));
#else
        uint4 nxt = {};
        if (hn)
            nxt = *(const uint4*)(wbf + (size_t)((t + 1) * 16 + srow) * DMODEL
                                      + schunk * 8);
#endif
        if (t + PFDIST < t1)
            __builtin_prefetch(wbf + (size_t)((t + PFDIST) * 16 + srow) * DMODEL
                                   + schunk * 8, 0, 1);   // global_prefetch_b8

        // Load ALL four B fragments into distinct registers first so the
        // scheduler can issue 8 ds_load_b128 back-to-back and hide their
        // latency under the WMMA chain with progressive s_wait_dscnt.
        const unsigned short* bb = smem + cur * LDS_BUF + l15 * LDS_ROW + kbB;
        v16bf bfrag[4];
        #pragma unroll
        for (int kf = 0; kf < 4; ++kf) {
            v8bf lo = *(const v8bf*)(bb + kf * 32);
            v8bf hi = *(const v8bf*)(bb + kf * 32 + 16);
            bfrag[kf] = __builtin_shufflevector(lo, hi, 0,1,2,3,4,5,6,7,
                                                        8,9,10,11,12,13,14,15);
        }

        // 4 K-frags x 2 row-tiles = 8 WMMAs (B shared per pair).
        v8f c0 = {}, c1 = {};
        #pragma unroll
        for (int kf = 0; kf < 4; ++kf) {
            c0 = __builtin_amdgcn_wmma_f32_16x16x32_bf16(
                     false, a0[kf], false, bfrag[kf], (short)0, c0, false, false);
            c1 = __builtin_amdgcn_wmma_f32_16x16x32_bf16(
                     false, a1[kf], false, bfrag[kf], (short)0, c1, false, false);
        }

        const float xmask = ((t * 16 + l15) < VOCAB) ? 0.f : -INFINITY;
        #pragma unroll
        for (int j = 0; j < 8; ++j) {
            lse_update(c0[j] + xmask, rm0[j], rs0[j]);
            lse_update(c1[j] + xmask, rm1[j], rs1[j]);
        }

#if USE_TDM
        if (w == 0)
            __builtin_amdgcn_s_wait_tensorcnt(0);      // next tile landed
#else
        if (hn)
            *(uint4*)&smem[(cur ^ 1) * LDS_BUF + loff] = nxt;
#endif
        __syncthreads();
    }

    // Merge across the 16 lanes of each half (columns), write row partials.
    #pragma unroll
    for (int tset = 0; tset < 2; ++tset) {
        #pragma unroll
        for (int j = 0; j < 8; ++j) {
            float mj = tset ? rm1[j] : rm0[j];
            float sj = tset ? rs1[j] : rs0[j];
            #pragma unroll
            for (int off = 8; off >= 1; off >>= 1) {
                float m2 = __shfl_xor(mj, off, 32);
                float s2 = __shfl_xor(sj, off, 32);
                float nm = fmaxf(mj, m2);
                sj = sj * __expf(mj - nm) + s2 * __expf(m2 - nm);
                mj = nm;
            }
            if (l15 == 0) {
                int grow = rbase + tset * 16 + (hi16 << 3) + j;
                partials[grow * NSPLIT + vsplit] = make_float2(mj, sj);
            }
        }
    }
}

// ---- K4: merge splits + center-token logit (f32) -> loss ------------------
__global__ void __launch_bounds__(256)
k_finalize(const float2* __restrict__ partials,
           const float* __restrict__ ctx_f32,
           const float* __restrict__ weight,
           const int* __restrict__ center_ids,
           float* __restrict__ out) {
    int b    = (blockIdx.x * 256 + threadIdx.x) >> 5;  // wave id == row
    int lane = threadIdx.x & 31;
    if (b >= BATCH) return;

    float2 p = partials[b * NSPLIT + lane];            // NSPLIT == 32 lanes
    float mj = p.x, sj = p.y;
    #pragma unroll
    for (int off = 16; off >= 1; off >>= 1) {
        float m2 = __shfl_xor(mj, off, 32);
        float s2 = __shfl_xor(sj, off, 32);
        float nm = fmaxf(mj, m2);
        sj = sj * __expf(mj - nm) + s2 * __expf(m2 - nm);
        mj = nm;
    }
    float logZ = mj + __logf(sj);

    int cid = center_ids[b];
    float acc = 0.f;
    #pragma unroll
    for (int i = 0; i < 4; ++i) {
        int d = lane + 32 * i;
        acc = fmaf(ctx_f32[b * DMODEL + d], weight[cid * DMODEL + d], acc);
    }
    #pragma unroll
    for (int off = 16; off >= 1; off >>= 1)
        acc += __shfl_xor(acc, off, 32);

    if (lane == 0) out[b] = logZ - acc;
}

// ---------------------------------------------------------------------------
extern "C" void kernel_launch(void* const* d_in, const int* in_sizes, int n_in,
                              void* d_out, int out_size, void* d_ws, size_t ws_size,
                              hipStream_t stream) {
    const float* emb        = (const float*)d_in[0];   // [VOCAB, 128]
    const float* weight     = (const float*)d_in[1];   // [VOCAB, 128]
    const int*   ctx_ids    = (const int*)d_in[2];     // [B, 10]
    const int*   center_ids = (const int*)d_in[3];     // [B]
    float*       out        = (float*)d_out;           // [B]

    char* ws = (char*)d_ws;
    float*          ctx_f32  = (float*)ws;                                 // 2 MB
    unsigned short* ctx_bf16 = (unsigned short*)(ws + (size_t)BATCH*DMODEL*4);   // 1 MB
    unsigned short* wbf      = (unsigned short*)(ws + (size_t)BATCH*DMODEL*6);   // 12.87 MB
    float2*         partials = (float2*)(ws + (size_t)BATCH*DMODEL*6
                                            + (size_t)VPAD*DMODEL*2);      // 1 MB

    k_ctx_mean<<<(BATCH * DMODEL + 255) / 256, 256, 0, stream>>>(
        emb, ctx_ids, ctx_f32, ctx_bf16);

    k_w_bf16<<<(VPAD * DMODEL + 255) / 256, 256, 0, stream>>>(weight, wbf);

    dim3 grid3(BATCH / ROWS_PER_WG, NSPLIT);           // (16, 32)
    k_gemm_lse<<<grid3, 256, 0, stream>>>(ctx_bf16, wbf, partials);

    k_finalize<<<BATCH / 8, 256, 0, stream>>>(partials, ctx_f32, weight,
                                              center_ids, out);
}
